// SDBlock_16338055594615
// MI455X (gfx1250) — matrix operand order repaired and checked by reading, not compiled, not run on hardware
//
#include <hip/hip_runtime.h>

// ---------------- problem constants ----------------
#define BB   4
#define CC   64
#define HH   128
#define WWp  128
#define GGn  4
#define CGc  16          // channels per group
#define KKt  9           // 3x3 taps
#define HWc  (HH*WWp)    // 16384
#define CHW  (CC*HWc)    // 1,048,576
#define NTOT (BB*CHW)    // 4,194,304
#define OCH  (2*GGn*KKt) // 72 offset channels
#define OPAD 80          // 72 -> padded M for 5x WMMA M-tiles
#define NOFF (BB*OCH*HWc)
#define EPSf 1e-5f
#define KPAD 160         // 144 -> padded K for 5x WMMA(K=32)

typedef __attribute__((ext_vector_type(16))) _Float16 v16h;
typedef __attribute__((ext_vector_type(8)))  float    v8f;

// ---------------- per-sample norm: stats ----------------
__global__ void gn_stats_kernel(const float* __restrict__ in,
                                float* __restrict__ stats) {
  __shared__ float ssum[256], ssq[256];
  int b   = blockIdx.x >> 6;        // 64 blocks per batch sample
  int blk = blockIdx.x & 63;
  const int chunk = CHW / 64;       // 16384
  const float* p = in + (size_t)b * CHW + (size_t)blk * chunk;
  float s = 0.f, q = 0.f;
  for (int i = threadIdx.x; i < chunk; i += 256) {
    float v = p[i]; s += v; q += v * v;
  }
  ssum[threadIdx.x] = s; ssq[threadIdx.x] = q;
  __syncthreads();
  for (int o = 128; o > 0; o >>= 1) {
    if ((int)threadIdx.x < o) {
      ssum[threadIdx.x] += ssum[threadIdx.x + o];
      ssq [threadIdx.x] += ssq [threadIdx.x + o];
    }
    __syncthreads();
  }
  if (threadIdx.x == 0) {
    atomicAdd(&stats[2 * b],     ssum[0]);
    atomicAdd(&stats[2 * b + 1], ssq [0]);
  }
}

// ---------------- norm apply (+optional relu) ----------------
__global__ void gn_apply_kernel(const float* __restrict__ in,
                                const float* __restrict__ stats,
                                const float* __restrict__ gamma,
                                const float* __restrict__ beta,
                                float* __restrict__ out, int do_relu) {
  int idx = blockIdx.x * 256 + threadIdx.x;
  if (idx >= NTOT) return;
  int b = idx >> 20;                 // /CHW (2^20)
  int c = (idx >> 14) & (CC - 1);
  float inv = 1.0f / (float)CHW;
  float m = stats[2 * b] * inv;
  float v = stats[2 * b + 1] * inv - m * m;
  float y = (in[idx] - m) * rsqrtf(v + EPSf) * gamma[c] + beta[c];
  if (do_relu) y = fmaxf(y, 0.0f);
  out[idx] = y;
}

// ---------------- depthwise 7x7 conv + bias ----------------
__global__ void dwconv_kernel(const float* __restrict__ xn,
                              const float* __restrict__ dww,
                              const float* __restrict__ dwb,
                              float* __restrict__ f) {
  int idx = blockIdx.x * 256 + threadIdx.x;
  if (idx >= NTOT) return;
  int w0 = idx & (WWp - 1);
  int h0 = (idx >> 7) & (HH - 1);
  int c  = (idx >> 14) & (CC - 1);
  const float* ch = xn + (size_t)(idx >> 14) * HWc;   // (b*CC + c) plane
  const float* wk = dww + c * 49;
  float acc = dwb[c];
  #pragma unroll
  for (int dy = 0; dy < 7; ++dy) {
    int y = h0 + dy - 3;
    if ((unsigned)y < (unsigned)HH) {
      #pragma unroll
      for (int dx = 0; dx < 7; ++dx) {
        int x = w0 + dx - 3;
        if ((unsigned)x < (unsigned)WWp) acc += wk[dy * 7 + dx] * ch[y * WWp + x];
      }
    }
  }
  f[idx] = acc;
}

// ---------------- pointwise 64 -> 72 offset head, WMMA ----------------
// Off[72, HW] = Wpw[72,64] @ F[64, HW] per batch sample; M padded to 80.
// grid: BB*5*1024/8 blocks of 256 threads (8 waves); wave -> one 16x16 tile.
__global__ void pwconv_wmma_kernel(const float* __restrict__ f,
                                   const float* __restrict__ pww,
                                   const float* __restrict__ pwb,
                                   float* __restrict__ off) {
  __shared__ _Float16 Alds[OPAD * CC];          // 80x64 f16 weights
  int wv = threadIdx.x >> 5, lane = threadIdx.x & 31;
  for (int i = threadIdx.x; i < OPAD * CC; i += 256) {
    int o = i >> 6; int c = i & (CC - 1);
    Alds[i] = (_Float16)((o < OCH) ? pww[o * CC + c] : 0.0f);
  }
  __syncthreads();

  int id  = blockIdx.x * 8 + wv;                // global tile id
  int b   = id / (5 * 1024);
  int rem = id % (5 * 1024);
  int mt  = rem >> 10;                          // 0..4 M-tile
  int pt  = rem & 1023;                         // pixel tile
  int p0  = pt * 16;
  int row = lane & 15, hi = lane >> 4;

  const float* fb = f + (size_t)b * CHW + p0 + row;
  v8f acc = {};
  #pragma unroll
  for (int t = 0; t < 2; ++t) {                 // K = 64 = 2 x 32
    v16h a, bf;
    #pragma unroll
    for (int i = 0; i < 8; ++i) {               // A 16x32 f16 layout
      a[i]     = Alds[(mt * 16 + row) * CC + t * 32 + hi * 8 + i];
      a[8 + i] = Alds[(mt * 16 + row) * CC + t * 32 + 16 + hi * 8 + i];
    }
    #pragma unroll
    for (int i = 0; i < 16; ++i)                // B 32x16 f16 layout
      bf[i] = (_Float16)fb[(size_t)(t * 32 + (i >> 3) * 16 + hi * 8 + (i & 7)) * HWc];
    acc = __builtin_amdgcn_wmma_f32_16x16x32_f16(
        false, a, false, bf, (short)0, acc, false, false);
  }
  #pragma unroll
  for (int r = 0; r < 8; ++r) {
    int o = mt * 16 + r + 8 * hi;
    if (o < OCH)
      off[((size_t)b * OCH + o) * HWc + p0 + row] = acc[r] + pwb[o];
  }
}

// ---------------- deformable sample + WMMA einsum (+bias +residual) ----------
// grid: BB*GGn*128 blocks, 256 threads (8 waves); each wave -> one 16x16 tile
template <bool RESID>
__global__ void deform_wmma_kernel(const float* __restrict__ xn,
                                   const float* __restrict__ off,
                                   const float* __restrict__ wconv, // (C,CG,3,3)
                                   const float* __restrict__ cbias,
                                   const float* __restrict__ resid,
                                   float* __restrict__ out) {
  __shared__ _Float16 Alds[16 * KPAD];          // weights A[m][kk]
  __shared__ _Float16 Vlds[8][16 * KPAD];       // per-wave val tile [n][kk]

  const int blocks_per_bg = 128;                // (HW/16)/8
  int bg = blockIdx.x / blocks_per_bg;
  int tb = blockIdx.x % blocks_per_bg;
  int b  = bg / GGn;
  int g  = bg % GGn;
  int wv   = threadIdx.x >> 5;
  int lane = threadIdx.x & 31;
  int tile = tb * 8 + wv;
  int p0   = tile * 16;

  // load A (16 x 144, zero-padded to 160) once per block
  for (int i = threadIdx.x; i < 16 * KPAD; i += 256) {
    int m  = i / KPAD;
    int kk = i - m * KPAD;
    float v = (kk < 144) ? wconv[(size_t)(g * 16 + m) * 144 + kk] : 0.0f;
    Alds[m * KPAD + kk] = (_Float16)v;
  }

  // zero the K-pad region of this wave's val tile
  _Float16* V = Vlds[wv];
  for (int i = lane; i < 16 * 16; i += 32) {
    int n  = i >> 4;
    int kk = 144 + (i & 15);
    V[n * KPAD + kk] = (_Float16)0.0f;
  }

  // bilinear sampling: 144 (k,n) pairs, 16 channels each
  const float* xg = xn + ((size_t)b * CC + (size_t)g * CGc) * HWc;
  __builtin_prefetch(xg, 0, 1);
  for (int pair = lane; pair < KKt * 16; pair += 32) {
    int k = pair >> 4;
    int n = pair & 15;
    int p  = p0 + n;
    int h0 = p >> 7;
    int w0 = p & (WWp - 1);
    int ki = k / 3, kj = k % 3;
    int chy = (g * KKt + k) * 2;
    float oy = off[((size_t)b * OCH + chy)     * HWc + p] * 1.0f; // OFF_SCALE
    float ox = off[((size_t)b * OCH + chy + 1) * HWc + p] * 1.0f;
    float cy = oy + (float)(h0 + ki - 1);
    float cx = ox + (float)(w0 + kj - 1);
    float y0f = floorf(cy), x0f = floorf(cx);
    float wy1 = cy - y0f, wx1 = cx - x0f;
    float wy0 = 1.0f - wy1, wx0 = 1.0f - wx1;
    int y0 = (int)y0f, x0 = (int)x0f;
    int y1 = y0 + 1,  x1 = x0 + 1;
    float v00 = ((unsigned)y0 < HH && (unsigned)x0 < WWp) ? wy0 * wx0 : 0.0f;
    float v01 = ((unsigned)y0 < HH && (unsigned)x1 < WWp) ? wy0 * wx1 : 0.0f;
    float v10 = ((unsigned)y1 < HH && (unsigned)x0 < WWp) ? wy1 * wx0 : 0.0f;
    float v11 = ((unsigned)y1 < HH && (unsigned)x1 < WWp) ? wy1 * wx1 : 0.0f;
    int yc0 = min(max(y0, 0), HH - 1), yc1 = min(max(y1, 0), HH - 1);
    int xc0 = min(max(x0, 0), WWp - 1), xc1 = min(max(x1, 0), WWp - 1);
    int i00 = yc0 * WWp + xc0, i01 = yc0 * WWp + xc1;
    int i10 = yc1 * WWp + xc0, i11 = yc1 * WWp + xc1;
    #pragma unroll
    for (int c = 0; c < CGc; ++c) {
      const float* ch = xg + (size_t)c * HWc;
      float v = v00 * ch[i00] + v01 * ch[i01] + v10 * ch[i10] + v11 * ch[i11];
      V[n * KPAD + (c * KKt + k)] = (_Float16)v;
    }
  }
  __syncthreads();

  // WMMA: acc[16x16] = A[16 x 160] * V^T[160 x 16]
  int row = lane & 15;
  int hi  = lane >> 4;
  v8f acc = {};
  #pragma unroll
  for (int t = 0; t < KPAD / 32; ++t) {
    v16h a, bf;
    #pragma unroll
    for (int i = 0; i < 8; ++i) {        // A 16x32 layout (16-bit)
      a[i]     = Alds[row * KPAD + t * 32 + hi * 8 + i];
      a[8 + i] = Alds[row * KPAD + t * 32 + 16 + hi * 8 + i];
    }
    #pragma unroll
    for (int i = 0; i < 16; ++i)         // B 32x16 layout (16-bit)
      bf[i] = V[row * KPAD + t * 32 + (i >> 3) * 16 + hi * 8 + (i & 7)];
    acc = __builtin_amdgcn_wmma_f32_16x16x32_f16(
        false, a, false, bf, (short)0, acc, false, false);
  }

  // D layout: VGPR r -> M = r + 8*hi, N = lane&15
  #pragma unroll
  for (int r = 0; r < 8; ++r) {
    int m  = r + 8 * hi;
    int oc = g * 16 + m;
    size_t oidx = ((size_t)b * CC + oc) * HWc + p0 + row;
    float v = acc[r] + cbias[oc];
    if (RESID) v += resid[oidx];
    out[oidx] = v;
  }
}

// ---------------- host-side orchestration ----------------
extern "C" void kernel_launch(void* const* d_in, const int* in_sizes, int n_in,
                              void* d_out, int out_size, void* d_ws, size_t ws_size,
                              hipStream_t stream) {
  const float* x    = (const float*)d_in[0];
  const float* gn_g = (const float*)d_in[1];
  const float* gn_b = (const float*)d_in[2];
  const float* dw_w = (const float*)d_in[3];
  const float* dw_b = (const float*)d_in[4];
  const float* pw_w = (const float*)d_in[5];
  const float* pw_b = (const float*)d_in[6];
  const float* wc   = (const float*)d_in[7];
  const float* bc   = (const float*)d_in[8];
  float* out = (float*)d_out;

  float* wsf   = (float*)d_ws;
  float* stats = wsf;                 // 2*BB floats (padded to 64)
  float* xn    = wsf + 64;
  float* f     = xn + NTOT;
  float* off   = f + NTOT;
  float* h1    = off + NOFF;
  float* hbuf  = h1 + NTOT;

  const float* stage_in[3]  = { x,  h1,  hbuf };
  float*       stage_out[3] = { h1, hbuf, out };
  const float* stage_res[3] = { nullptr, x, hbuf };

  for (int s = 0; s < 3; ++s) {
    const float* in = stage_in[s];
    hipMemsetAsync(stats, 0, 2 * BB * sizeof(float), stream);
    gn_stats_kernel<<<BB * 64, 256, 0, stream>>>(in, stats);
    gn_apply_kernel<<<NTOT / 256, 256, 0, stream>>>(
        in, stats, gn_g + s * CC, gn_b + s * CC, xn, (s < 2) ? 1 : 0);
    dwconv_kernel<<<NTOT / 256, 256, 0, stream>>>(
        xn, dw_w + s * CC * 49, dw_b + s * CC, f);
    pwconv_wmma_kernel<<<BB * 5 * 1024 / 8, 256, 0, stream>>>(
        f, pw_w + s * OCH * CC, pw_b + s * OCH, off);
    if (s == 0)
      deform_wmma_kernel<false><<<BB * GGn * 128, 256, 0, stream>>>(
          xn, off, wc + s * CC * CGc * 9, bc + s * CC, nullptr, stage_out[s]);
    else
      deform_wmma_kernel<true><<<BB * GGn * 128, 256, 0, stream>>>(
          xn, off, wc + s * CC * CGc * 9, bc + s * CC, stage_res[s], stage_out[s]);
  }
}
